// AdditiveAttention_66374424593081
// MI455X (gfx1250) — matrix operand order repaired and checked by reading, not compiled, not run on hardware
//
#include <hip/hip_runtime.h>
#include <hip/hip_bf16.h>
#include <math.h>

// ---- vector types for WMMA ----
typedef __attribute__((ext_vector_type(16))) _Float16 v16h;
typedef __attribute__((ext_vector_type(8)))  _Float16 v8h;
typedef __attribute__((ext_vector_type(8)))  float    v8f;

#define Lq 512
#define Dd 768

// ---------------------------------------------------------------------------
// Elementwise f32 -> f16 convert
// ---------------------------------------------------------------------------
__global__ void cvt_f32_f16(const float* __restrict__ in, _Float16* __restrict__ out, int n) {
    int i = blockIdx.x * 256 + threadIdx.x;
    if (i < n) out[i] = (_Float16)in[i];
}

// ---------------------------------------------------------------------------
// Transpose [R,C] f32 -> [C,R] f16 (R,C multiples of 16), LDS-tiled coalesced
// ---------------------------------------------------------------------------
__global__ void transpose_f32_f16(const float* __restrict__ in, _Float16* __restrict__ out,
                                  int R, int C) {
    __shared__ float tile[16][17];
    int r = blockIdx.y * 16 + threadIdx.y;
    int c = blockIdx.x * 16 + threadIdx.x;
    tile[threadIdx.y][threadIdx.x] = in[r * C + c];
    __syncthreads();
    int orow = blockIdx.x * 16 + threadIdx.y;
    int ocol = blockIdx.y * 16 + threadIdx.x;
    out[orow * R + ocol] = (_Float16)tile[threadIdx.x][threadIdx.y];
}

// ---------------------------------------------------------------------------
// WMMA GEMM: C[M,N] = A[M,K] * Bt[N,K]^T (+ bias[n]) (+ addend[m,n])
// One wave (32 threads) computes one 16x16 tile, looping K in steps of 32.
// A/Bt are f16 row-major, accumulation f32 via v_wmma_f32_16x16x32_f16.
// ---------------------------------------------------------------------------
__global__ void __launch_bounds__(32)
wmma_gemm_f16(const _Float16* __restrict__ A, const _Float16* __restrict__ Bt,
              const float* __restrict__ bias, const float* __restrict__ addend,
              float* __restrict__ C, int M, int N, int Kdim) {
    const int tileM  = blockIdx.x * 16;
    const int tileN  = blockIdx.y * 16;
    const int lane   = threadIdx.x & 31;
    const int lane16 = lane & 15;
    const int hlf    = lane >> 4;       // 0 or 1: which lane half
    const int koff   = hlf * 8;         // A-matrix K sub-offset per ISA layout

    const _Float16* arow = A  + (size_t)(tileM + lane16) * Kdim;
    const _Float16* brow = Bt + (size_t)(tileN + lane16) * Kdim;

    v8f acc = {};
    for (int k0 = 0; k0 < Kdim; k0 += 32) {
        union { v16h v; v8h h[2]; } a, b;
        // A 16x32 f16: halves 0-7 -> K = k0+koff.., halves 8-15 -> K = k0+16+koff..
        a.h[0] = *(const v8h*)(arow + k0 + koff);
        a.h[1] = *(const v8h*)(arow + k0 + 16 + koff);
        // B 32x16 f16: same K order as A: halves 0-7 -> K = k0+koff.., halves 8-15 -> K = k0+16+koff..
        b.h[0] = *(const v8h*)(brow + k0 + koff);
        b.h[1] = *(const v8h*)(brow + k0 + 16 + koff);
        acc = __builtin_amdgcn_wmma_f32_16x16x32_f16(
            /*neg_a=*/false, a.v, /*neg_b=*/false, b.v,
            /*c_mod=*/(short)0, acc, /*reuse_a=*/false, /*reuse_b=*/false);
    }

    const int col = tileN + lane16;
    const float bv = bias ? bias[col] : 0.0f;
    #pragma unroll
    for (int r = 0; r < 8; ++r) {
        const int row = tileM + hlf * 8 + r;   // C/D layout: VGPR r -> M = r + 8*half
        float v = acc[r] + bv;
        if (addend) v += addend[(size_t)row * N + col];
        C[(size_t)row * N + col] = v;
    }
}

// ---------------------------------------------------------------------------
// scores[q,k] = sum_d tanh(Q[q,d] + K[k,d]) * w[d] + b + mask[k]
// 16x16 (q,k) tile per 256-thread block; Q/K/w staged in padded LDS.
// This is the dominant cost (512^2 x 768 tanh-FMAs) and is VALU-bound.
// ---------------------------------------------------------------------------
#define DCH 64
__global__ void __launch_bounds__(256)
scores_kernel(const float* __restrict__ Q, const float* __restrict__ K,
              const float* __restrict__ w_att, const float* __restrict__ b_att,
              const float* __restrict__ mask, float* __restrict__ scores) {
    __shared__ float sQ[16][DCH + 1];
    __shared__ float sK[16][DCH + 1];
    __shared__ float sw[DCH];
    const int tk = threadIdx.x, tq = threadIdx.y;
    const int tid = tq * 16 + tk;
    const int qbase = blockIdx.y * 16, kbase = blockIdx.x * 16;

    float acc = 0.0f;
    for (int d0 = 0; d0 < Dd; d0 += DCH) {
        for (int i = tid; i < 16 * DCH; i += 256) {
            int rr = i / DCH, dd = i % DCH;
            sQ[rr][dd] = Q[(size_t)(qbase + rr) * Dd + d0 + dd];
            sK[rr][dd] = K[(size_t)(kbase + rr) * Dd + d0 + dd];
        }
        if (tid < DCH) sw[tid] = w_att[d0 + tid];
        __syncthreads();
        #pragma unroll 8
        for (int dd = 0; dd < DCH; ++dd)
            acc = fmaf(tanhf(sQ[tq][dd] + sK[tk][dd]), sw[dd], acc);
        __syncthreads();
    }
    const int q = qbase + tq, k = kbase + tk;
    scores[(size_t)q * Lq + k] = acc + b_att[0] + mask[k];
}

// ---------------------------------------------------------------------------
// Row softmax over 512 cols; writes normalized probs back (f32) + f16 copy.
// One 256-thread block per row.
// ---------------------------------------------------------------------------
__global__ void __launch_bounds__(256)
softmax_kernel(float* __restrict__ scores, _Float16* __restrict__ probsH) {
    __shared__ float red[256];
    const int q = blockIdx.x, tid = threadIdx.x;
    float* row = scores + (size_t)q * Lq;

    float m = -INFINITY;
    for (int k = tid; k < Lq; k += 256) m = fmaxf(m, row[k]);
    red[tid] = m; __syncthreads();
    for (int s = 128; s > 0; s >>= 1) { if (tid < s) red[tid] = fmaxf(red[tid], red[tid + s]); __syncthreads(); }
    m = red[0]; __syncthreads();

    float e0 = expf(row[tid]       - m);
    float e1 = expf(row[tid + 256] - m);
    red[tid] = e0 + e1; __syncthreads();
    for (int s = 128; s > 0; s >>= 1) { if (tid < s) red[tid] += red[tid + s]; __syncthreads(); }
    const float inv = 1.0f / red[0];

    float p0 = e0 * inv, p1 = e1 * inv;
    row[tid]       = p0;  probsH[(size_t)q * Lq + tid]       = (_Float16)p0;
    row[tid + 256] = p1;  probsH[(size_t)q * Lq + tid + 256] = (_Float16)p1;
}

// ---------------------------------------------------------------------------
extern "C" void kernel_launch(void* const* d_in, const int* in_sizes, int n_in,
                              void* d_out, int out_size, void* d_ws, size_t ws_size,
                              hipStream_t stream) {
    const float* hs    = (const float*)d_in[0];
    const float* mask  = (const float*)d_in[1];
    const float* Wq    = (const float*)d_in[2];
    const float* bq    = (const float*)d_in[3];
    const float* Wk    = (const float*)d_in[4];
    const float* bk    = (const float*)d_in[5];
    const float* w_att = (const float*)d_in[6];
    const float* b_att = (const float*)d_in[7];
    const float* Wt    = (const float*)d_in[8];
    const float* bt    = (const float*)d_in[9];
    float* out = (float*)d_out;

    char* ws = (char*)d_ws;
    size_t off = 0;
    auto alloc = [&](size_t bytes) { void* p = ws + off; off += (bytes + 255) & ~(size_t)255; return p; };

    _Float16* hsH       = (_Float16*)alloc((size_t)Lq * Dd * 2);  // hs f16           [512,768]
    _Float16* hsTH      = (_Float16*)alloc((size_t)Dd * Lq * 2);  // hs^T f16         [768,512]
    _Float16* WqT       = (_Float16*)alloc((size_t)Dd * Dd * 2);  // Wq^T f16         [768,768]
    _Float16* WkT       = (_Float16*)alloc((size_t)Dd * Dd * 2);
    _Float16* WtT       = (_Float16*)alloc((size_t)Dd * Dd * 2);
    float*    Qb        = (float*)   alloc((size_t)Lq * Dd * 4);  // Q f32            [512,768]
    float*    Kb        = (float*)   alloc((size_t)Lq * Dd * 4);
    float*    scores    = (float*)   alloc((size_t)Lq * Lq * 4);  // scores -> probs  [512,512]
    _Float16* probsH    = (_Float16*)alloc((size_t)Lq * Lq * 2);
    float*    weighted  = (float*)   alloc((size_t)Lq * Dd * 4);
    _Float16* weightedH = (_Float16*)alloc((size_t)Lq * Dd * 2);

    // --- prep: f16 conversions / weight transposes ---
    cvt_f32_f16<<<(Lq * Dd + 255) / 256, 256, 0, stream>>>(hs, hsH, Lq * Dd);
    transpose_f32_f16<<<dim3(Dd / 16, Lq / 16), dim3(16, 16), 0, stream>>>(hs, hsTH, Lq, Dd);
    transpose_f32_f16<<<dim3(Dd / 16, Dd / 16), dim3(16, 16), 0, stream>>>(Wq, WqT, Dd, Dd);
    transpose_f32_f16<<<dim3(Dd / 16, Dd / 16), dim3(16, 16), 0, stream>>>(Wk, WkT, Dd, Dd);
    transpose_f32_f16<<<dim3(Dd / 16, Dd / 16), dim3(16, 16), 0, stream>>>(Wt, WtT, Dd, Dd);

    // --- Q = hs@Wq + bq ; K = hs@Wk + bk (WMMA) ---
    wmma_gemm_f16<<<dim3(Lq / 16, Dd / 16), 32, 0, stream>>>(hsH, WqT, bq, nullptr, Qb, Lq, Dd, Dd);
    wmma_gemm_f16<<<dim3(Lq / 16, Dd / 16), 32, 0, stream>>>(hsH, WkT, bk, nullptr, Kb, Lq, Dd, Dd);

    // --- additive-attention score map (tanh, VALU-bound) ---
    scores_kernel<<<dim3(Lq / 16, Lq / 16), dim3(16, 16), 0, stream>>>(Qb, Kb, w_att, b_att, mask, scores);

    // --- softmax over k ---
    softmax_kernel<<<Lq, 256, 0, stream>>>(scores, probsH);

    // --- weighted = probs @ hs (WMMA, K=512) ---
    wmma_gemm_f16<<<dim3(Lq / 16, Dd / 16), 32, 0, stream>>>(probsH, hsTH, nullptr, nullptr, weighted, Lq, Dd, Lq);
    cvt_f32_f16<<<(Lq * Dd + 255) / 256, 256, 0, stream>>>(weighted, weightedH, Lq * Dd);

    // --- out = weighted @ Wt + bt + Q (WMMA + residual epilogue) ---
    wmma_gemm_f16<<<dim3(Lq / 16, Dd / 16), 32, 0, stream>>>(weightedH, WtT, bt, Qb, out, Lq, Dd, Dd);
}
